// HeteroGNN_44074954392266
// MI455X (gfx1250) — compile-verified, matrix operand order checked by reading, not run on hardware
//
#include <hip/hip_runtime.h>
#include <math.h>

// ---------------- constants (match reference) ----------------
#define NN   100000      // nodes per node type
#define HID  128
#define NB   256         // batch size B
#define NLAY 2
#define SLABS 5          // 16-row slabs per block in the WMMA GEMM (100000 = 1250*5*16)

typedef __attribute__((ext_vector_type(16))) __bf16 v16bf;
typedef __attribute__((ext_vector_type(8)))  float  v8f;

// ---------------- helpers ----------------
__device__ __forceinline__ unsigned short f2bf_raw(float f) {
    unsigned u = __float_as_uint(f);
    unsigned r = u + 0x7FFFu + ((u >> 16) & 1u);   // RNE
    return (unsigned short)(r >> 16);
}

__device__ __forceinline__ void atomicMaxF(float* addr, float v) {
    // classic two-path trick: monotone int for v>=0, monotone-reversed uint for v<0
    if (v >= 0.0f) atomicMax((int*)addr, __float_as_int(v));
    else           atomicMin((unsigned int*)addr, __float_as_uint(v));
}

// ---------------- fill ----------------
__global__ void k_fill(float* __restrict__ p, float v, long long n) {
    long long i = (long long)blockIdx.x * blockDim.x + threadIdx.x;
    if (i < n) p[i] = v;
}

// ---------------- f32 -> bf16 convert ----------------
__global__ void k_cvt_bf16(const float* __restrict__ src,
                           unsigned short* __restrict__ dst, long long n) {
    long long i = (long long)blockIdx.x * blockDim.x + threadIdx.x;
    if (i < n) dst[i] = f2bf_raw(src[i]);
}

// ---------------- pack one 128x128 weight (row-major [k][n], f32)
// into WMMA-native bf16 B layout: Wp[((ks*8+nt)*32+lane)*16 + i]
// holds W[k = ks*32 + (i/8)*16 + (lane/16)*8 + (i%8)][n = nt*16 + lane%16]
__global__ void k_pack_w(const float* __restrict__ W, unsigned short* __restrict__ Wp) {
    int idx = blockIdx.x * blockDim.x + threadIdx.x;   // 0..16383
    if (idx >= 128 * 128) return;
    int i    = idx & 15;
    int lane = (idx >> 4) & 31;
    int nt   = (idx >> 9) & 7;
    int ks   = (idx >> 12) & 3;
    int k = ks * 32 + (i >> 3) * 16 + (lane >> 4) * 8 + (i & 7);
    int n = nt * 16 + (lane & 15);
    Wp[idx] = f2bf_raw(W[k * 128 + n]);
}

// ---------------- WMMA GEMM: C[N,128] = A[N,128](bf16) @ W(packed bf16) ----
// block = 256 threads = 8 waves; wave w -> 16-col tile.
// Each block handles SLABS consecutive 16-row slabs, holding the B (weight)
// operand in registers across slabs: 20 v_wmma per B fetch instead of 4.
__global__ void k_gemm_wmma(const unsigned short* __restrict__ A,
                            const unsigned short* __restrict__ Wp,
                            float* __restrict__ C, int nrows) {
    int wave = threadIdx.x >> 5;
    int lane = threadIdx.x & 31;
    int half = lane >> 4;
    int rowbase0 = blockIdx.x * (16 * SLABS);

    union BV { float4 f[2]; v16bf v; };
    BV b[4];
#pragma unroll
    for (int ks = 0; ks < 4; ++ks) {   // hoisted: weight operand lives in VGPRs
        const float4* Bp = (const float4*)(Wp + (((ks * 8 + wave) * 32 + lane) * 16));
        b[ks].f[0] = Bp[0];
        b[ks].f[1] = Bp[1];
    }

    int col = wave * 16 + (lane & 15);
#pragma unroll
    for (int sl = 0; sl < SLABS; ++sl) {
        int rowbase = rowbase0 + sl * 16;
        if (rowbase >= nrows) break;
        int row = rowbase + (lane & 15);
        const float4* Arow = (const float4*)(A + (size_t)row * HID);
        v8f acc = {};
#pragma unroll
        for (int ks = 0; ks < 4; ++ks) {
            BV a;
            a.f[0] = Arow[ks * 4 + half];       // K = ks*32 + half*8      .. +8
            a.f[1] = Arow[ks * 4 + 2 + half];   // K = ks*32 + 16 + half*8 .. +8
            acc = __builtin_amdgcn_wmma_f32_16x16x32_bf16(
                false, a.v, false, b[ks].v, (short)0, acc, false, false);
        }
        float* Cp = C + (size_t)(rowbase + half * 8) * HID + col;
#pragma unroll
        for (int r = 0; r < 8; ++r) Cp[(size_t)r * HID] = acc[r];  // VGPR r -> row half*8+r
    }
}

// ---------------- per-row dot: out[i] = H[i,:] . a  (wave per row) --------
__global__ void k_rowdot(const float* __restrict__ H, const float* __restrict__ a,
                         float* __restrict__ out, int n) {
    int wid  = (int)(((long long)blockIdx.x * blockDim.x + threadIdx.x) >> 5);
    int lane = threadIdx.x & 31;
    if (wid >= n) return;
    float4 hv = ((const float4*)(H + (size_t)wid * HID))[lane];
    float4 av = ((const float4*)a)[lane];
    float s = hv.x * av.x + hv.y * av.y + hv.z * av.z + hv.w * av.w;
#pragma unroll
    for (int off = 16; off > 0; off >>= 1) s += __shfl_down(s, off, 32);
    if (lane == 0) out[wid] = s;
}

// ---------------- edge scores + segment max --------------------------------
__global__ void k_edge_scores(const int* __restrict__ src, const int* __restrict__ dst,
                              const float* __restrict__ ss, const float* __restrict__ sd,
                              float* __restrict__ es, float* __restrict__ m, int ne) {
    int e = blockIdx.x * blockDim.x + threadIdx.x;
    if (e >= ne) return;
    float v = ss[src[e]] + sd[dst[e]];
    v = (v > 0.0f) ? v : 0.2f * v;                 // leaky_relu(0.2)
    es[e] = v;
    atomicMaxF(&m[dst[e]], v);
}

// ---------------- exp(e - m[dst]) + segment sum ----------------------------
__global__ void k_edge_exp(const int* __restrict__ dst, const float* __restrict__ es,
                           const float* __restrict__ m, float* __restrict__ pe,
                           float* __restrict__ den, int ne) {
    int e = blockIdx.x * blockDim.x + threadIdx.x;
    if (e >= ne) return;
    int d = dst[e];
    float p = expf(es[e] - m[d]);
    pe[e] = p;
    atomicAdd(&den[d], p);
}

// ---------------- scatter: acc[dst] += alpha * hs[src]  (wave per edge) ----
__global__ void k_scatter(const int* __restrict__ src, const int* __restrict__ dst,
                          const float* __restrict__ pe, const float* __restrict__ den,
                          const float* __restrict__ hs, float* __restrict__ acc, int ne) {
    int gw   = (int)(((long long)blockIdx.x * blockDim.x + threadIdx.x) >> 5);
    int lane = threadIdx.x & 31;
    int nw   = (int)(((long long)gridDim.x * blockDim.x) >> 5);
    for (int e = gw; e < ne; e += nw) {
        int s = src[e], d = dst[e];
        if (e + nw < ne)  // gfx1250 global_prefetch_b8 on next edge's feature row
            __builtin_prefetch(hs + (size_t)src[e + nw] * HID + lane * 4, 0, 1);
        float alpha = pe[e] / fmaxf(den[d], 1e-16f);
        float4 hv = ((const float4*)(hs + (size_t)s * HID))[lane];
        float* ap = acc + (size_t)d * HID + lane * 4;
        atomicAdd(ap + 0, alpha * hv.x);
        atomicAdd(ap + 1, alpha * hv.y);
        atomicAdd(ap + 2, alpha * hv.z);
        atomicAdd(ap + 3, alpha * hv.w);
    }
}

// ---------------- bias + tanh -> f32 & bf16 next features ------------------
__global__ void k_finish(const float* __restrict__ acc, const float* __restrict__ b1,
                         const float* __restrict__ b2, float* __restrict__ xf,
                         unsigned short* __restrict__ xb, long long n) {
    long long i = (long long)blockIdx.x * blockDim.x + threadIdx.x;
    if (i >= n) return;
    int col = (int)(i & (HID - 1));
    float v = tanhf(acc[i] + b1[col] + b2[col]);
    xf[i] = v;
    xb[i] = f2bf_raw(v);
}

// ---------------- mean-pool accumulation (wave per node) -------------------
__global__ void k_pool(const float* __restrict__ x, const int* __restrict__ batch,
                       float* __restrict__ psum, float* __restrict__ cnt, int n) {
    int wid  = (int)(((long long)blockIdx.x * blockDim.x + threadIdx.x) >> 5);
    int lane = threadIdx.x & 31;
    if (wid >= n) return;
    int b = batch[wid];
    float4 hv = ((const float4*)(x + (size_t)wid * HID))[lane];
    float* pp = psum + (size_t)b * HID + lane * 4;
    atomicAdd(pp + 0, hv.x);
    atomicAdd(pp + 1, hv.y);
    atomicAdd(pp + 2, hv.z);
    atomicAdd(pp + 3, hv.w);
    if (lane == 0) atomicAdd(&cnt[b], 1.0f);
}

// ---------------- head: sigmoid(0.5*(pi+pj) @ lin_w + lin_b) (wave per b) --
__global__ void k_head(const float* __restrict__ pool, const float* __restrict__ cnt,
                       const float* __restrict__ lw, const float* __restrict__ lb,
                       float* __restrict__ out) {
    int wid  = (int)(((long long)blockIdx.x * blockDim.x + threadIdx.x) >> 5);
    int lane = threadIdx.x & 31;
    if (wid >= NB) return;
    float ci = fmaxf(cnt[wid], 1.0f);
    float cj = fmaxf(cnt[NB + wid], 1.0f);
    float4 pi = ((const float4*)(pool + (size_t)wid * HID))[lane];
    float4 pj = ((const float4*)(pool + (size_t)(NB + wid) * HID))[lane];
    float4 w  = ((const float4*)lw)[lane];
    float s = 0.5f * ((pi.x / ci + pj.x / cj) * w.x + (pi.y / ci + pj.y / cj) * w.y +
                      (pi.z / ci + pj.z / cj) * w.z + (pi.w / ci + pj.w / cj) * w.w);
#pragma unroll
    for (int off = 16; off > 0; off >>= 1) s += __shfl_down(s, off, 32);
    if (lane == 0) out[wid] = 1.0f / (1.0f + expf(-(s + lb[0])));
}

// ==========================================================================
extern "C" void kernel_launch(void* const* d_in, const int* in_sizes, int n_in,
                              void* d_out, int out_size, void* d_ws, size_t ws_size,
                              hipStream_t stream) {
    const float* x_i  = (const float*)d_in[0];
    const float* x_j  = (const float*)d_in[1];
    const int*   ei[4] = { (const int*)d_in[2], (const int*)d_in[3],
                           (const int*)d_in[4], (const int*)d_in[5] }; // ii, jj, ij, ji
    const int*   batch_i = (const int*)d_in[6];
    const int*   batch_j = (const int*)d_in[7];
    const float* Ws   = (const float*)d_in[8];   // [2,4,128,128]
    const float* Wd   = (const float*)d_in[9];
    const float* atts = (const float*)d_in[10];  // [2,4,128]
    const float* attd = (const float*)d_in[11];
    const float* bias = (const float*)d_in[12];  // [2,4,128]
    const float* lw   = (const float*)d_in[13];  // [128,1]
    const float* lb   = (const float*)d_in[14];  // [1]
    const int E = in_sizes[2] / 2;

    // ---- carve workspace (256B aligned) ----
    char* wp = (char*)d_ws;
    auto carve = [&](size_t bytes) -> void* {
        void* p = (void*)wp;
        wp += (bytes + 255) & ~(size_t)255;
        return p;
    };
    const long long NF = (long long)NN * HID;
    float*          xi_f  = (float*)carve(NF * 4);
    float*          xj_f  = (float*)carve(NF * 4);
    unsigned short* xi_b  = (unsigned short*)carve(NF * 2);
    unsigned short* xj_b  = (unsigned short*)carve(NF * 2);
    float*          acc_i = (float*)carve(NF * 4);
    float*          acc_j = (float*)carve(NF * 4);
    float*          hbuf  = (float*)carve(NF * 4);
    unsigned short* Wb    = (unsigned short*)carve((size_t)128 * 128 * 2);
    float*          sc_s  = (float*)carve((size_t)NN * 4);
    float*          sc_d  = (float*)carve((size_t)NN * 4);
    float*          mseg  = (float*)carve((size_t)NN * 4);
    float*          den   = (float*)carve((size_t)NN * 4);
    float*          es    = (float*)carve((size_t)E * 4);
    float*          pe    = (float*)carve((size_t)E * 4);
    float*          pool  = (float*)carve((size_t)2 * NB * HID * 4);
    float*          cnt   = (float*)carve((size_t)2 * NB * 4);

    const int TB = 256;
    const int gElemNF   = (int)((NF + TB - 1) / TB);         // elementwise over N*128
    const int gElemN    = (NN + TB - 1) / TB;                // elementwise over N
    const int gElemE    = (E + TB - 1) / TB;                 // elementwise over E
    const int gWaveN    = (int)(((long long)NN * 32 + TB - 1) / TB); // wave per node
    const int gWaveE    = (int)(((long long)E * 32 + TB - 1) / TB);  // wave per edge
    const int gGemm     = (NN + 16 * SLABS - 1) / (16 * SLABS);      // 1250 blocks
    const int gPack     = (128 * 128 + TB - 1) / TB;

    // seed bf16 features from inputs (layer 0)
    k_cvt_bf16<<<gElemNF, TB, 0, stream>>>(x_i, xi_b, NF);
    k_cvt_bf16<<<gElemNF, TB, 0, stream>>>(x_j, xj_b, NF);

    for (int l = 0; l < NLAY; ++l) {
        k_fill<<<gElemNF, TB, 0, stream>>>(acc_i, 0.0f, NF);
        k_fill<<<gElemNF, TB, 0, stream>>>(acc_j, 0.0f, NF);

        // (weight type t, src bf16, dst bf16, edges, accumulator)
        struct Conv { int t; const unsigned short* sb; const unsigned short* db;
                      const int* e; float* acc; };
        const Conv convs[4] = {
            { 0, xi_b, xi_b, ei[0], acc_i },   // inner_i
            { 3, xj_b, xi_b, ei[3], acc_i },   // outer_ji -> i
            { 1, xj_b, xj_b, ei[1], acc_j },   // inner_j
            { 2, xi_b, xj_b, ei[2], acc_j },   // outer_ij -> j
        };
        for (int c = 0; c < 4; ++c) {
            const Conv& cv = convs[c];
            const size_t wo = ((size_t)l * 4 + cv.t) * 128 * 128;
            const size_t ao = ((size_t)l * 4 + cv.t) * 128;
            const int* srcI = cv.e;
            const int* dstI = cv.e + E;

            // hd = x_dst @ Wd  -> reduce to per-node dst score
            k_pack_w   <<<gPack,  TB, 0, stream>>>(Wd + wo, Wb);
            k_gemm_wmma<<<gGemm,  TB, 0, stream>>>(cv.db, Wb, hbuf, NN);
            k_rowdot   <<<gWaveN, TB, 0, stream>>>(hbuf, attd + ao, sc_d, NN);
            // hs = x_src @ Ws  -> per-node src score; keep hs for scatter
            k_pack_w   <<<gPack,  TB, 0, stream>>>(Ws + wo, Wb);
            k_gemm_wmma<<<gGemm,  TB, 0, stream>>>(cv.sb, Wb, hbuf, NN);
            k_rowdot   <<<gWaveN, TB, 0, stream>>>(hbuf, atts + ao, sc_s, NN);

            // segment softmax over edges
            k_fill<<<gElemN, TB, 0, stream>>>(mseg, -INFINITY, NN);
            k_fill<<<gElemN, TB, 0, stream>>>(den, 0.0f, NN);
            k_edge_scores<<<gElemE, TB, 0, stream>>>(srcI, dstI, sc_s, sc_d, es, mseg, E);
            k_edge_exp   <<<gElemE, TB, 0, stream>>>(dstI, es, mseg, pe, den, E);
            k_scatter    <<<gWaveE, TB, 0, stream>>>(srcI, dstI, pe, den, hbuf, cv.acc, E);
        }

        // out_i: biases of types 0 and 3; out_j: types 1 and 2 ; tanh -> next features
        k_finish<<<gElemNF, TB, 0, stream>>>(acc_i, bias + ((size_t)l * 4 + 0) * 128,
                                             bias + ((size_t)l * 4 + 3) * 128,
                                             xi_f, xi_b, NF);
        k_finish<<<gElemNF, TB, 0, stream>>>(acc_j, bias + ((size_t)l * 4 + 1) * 128,
                                             bias + ((size_t)l * 4 + 2) * 128,
                                             xj_f, xj_b, NF);
    }

    // mean pool + head
    k_fill<<<(2 * NB * HID + TB - 1) / TB, TB, 0, stream>>>(pool, 0.0f, 2 * NB * HID);
    k_fill<<<(2 * NB + TB - 1) / TB, TB, 0, stream>>>(cnt, 0.0f, 2 * NB);
    k_pool<<<gWaveN, TB, 0, stream>>>(xi_f, batch_i, pool, cnt, NN);
    k_pool<<<gWaveN, TB, 0, stream>>>(xj_f, batch_j, pool + (size_t)NB * HID, cnt + NB, NN);
    k_head<<<(NB * 32 + TB - 1) / TB, TB, 0, stream>>>(pool, cnt, lw, lb, (float*)d_out);
}
